// Attention_71871982731739
// MI455X (gfx1250) — matrix operand order repaired and checked by reading, not compiled, not run on hardware
//
#include <hip/hip_runtime.h>
#include <stdint.h>

// ---------------------------------------------------------------------------
// Types for CDNA5 WMMA (wave32): D(16x16,f32) = A(16x32,bf16) * B(32x16,bf16) + C
// ---------------------------------------------------------------------------
typedef __attribute__((ext_vector_type(16))) __bf16 v16bf;
typedef __attribute__((ext_vector_type(2)))  __bf16 v2bf;
typedef __attribute__((ext_vector_type(8)))  float  v8f;
typedef __attribute__((ext_vector_type(4)))  unsigned int u32x4;
typedef __attribute__((ext_vector_type(8)))  int i32x8;
typedef __attribute__((ext_vector_type(4)))  int i32x4;
typedef __attribute__((ext_vector_type(4)))  int v4i;

union FragAB {
    uint4 u[2];   // 2 x 16B = 8 VGPRs = 16 bf16
    v16bf v;
};

__device__ __forceinline__ unsigned short f2bf(float f) {
    unsigned int u = __float_as_uint(f);
    u += 0x7FFFu + ((u >> 16) & 1u);      // round-to-nearest-even
    return (unsigned short)(u >> 16);
}

// two f32 -> packed bf16 pair in one VALU op when available
__device__ __forceinline__ unsigned pack_bf16(float lo, float hi) {
#if __has_builtin(__builtin_amdgcn_cvt_pk_bf16_f32)
    union { v2bf v; unsigned u; } cv;
    cv.v = __builtin_amdgcn_cvt_pk_bf16_f32(lo, hi);
    return cv.u;
#else
    return (unsigned)f2bf(lo) | ((unsigned)f2bf(hi) << 16);
#endif
}

__device__ __forceinline__ v8f wmma_bf16(const FragAB& a, const FragAB& b, v8f c) {
    return __builtin_amdgcn_wmma_f32_16x16x32_bf16(
        false, a.v, false, b.v, (short)0, c, false, false);
}

__device__ __forceinline__ float fast_exp2(float x) {
#if __has_builtin(__builtin_amdgcn_exp2f)
    return __builtin_amdgcn_exp2f(x);     // native v_exp_f32 (base 2)
#else
    return exp2f(x);
#endif
}

// ---------------------------------------------------------------------------
// CDNA5 async copy to LDS (ASYNCcnt)
// ---------------------------------------------------------------------------
__device__ __forceinline__ void async_copy16(void* lds, const void* g) {
#if __has_builtin(__builtin_amdgcn_global_load_async_to_lds_b128)
    __builtin_amdgcn_global_load_async_to_lds_b128(
        (__attribute__((address_space(1))) v4i*)(uintptr_t)g,
        (__attribute__((address_space(3))) v4i*)(unsigned)(uintptr_t)lds,
        0, 0);
#else
    unsigned l = (unsigned)(uintptr_t)lds;
    unsigned long long ga = (unsigned long long)(uintptr_t)g;
    asm volatile("global_load_async_to_lds_b128 %0, %1, off"
                 :: "v"(l), "v"(ga) : "memory");
#endif
}

__device__ __forceinline__ void wait_async0() {
#if __has_builtin(__builtin_amdgcn_s_wait_asynccnt)
    __builtin_amdgcn_s_wait_asynccnt(0);
#else
    asm volatile("s_wait_asynccnt 0x0" ::: "memory");
#endif
}

// ---------------------------------------------------------------------------
// CDNA5 Tensor Data Mover. Group-1 (dims/strides/pad) built once, hoisted out
// of loops; per-issue only lds_addr + global_addr words change.
// LDS pad: interval code 4 (32 DWORDs=128B), amount code 3 (4 DWORDs=16B)
// -> effective LDS row stride 144B = 72 bf16.
// ---------------------------------------------------------------------------
#if __has_builtin(__builtin_amdgcn_tensor_load_to_lds)
#define HAVE_TDM 1
#if __has_include(<hip/amd_detail/amd_gfx1250_TDM.h>)
#define TDM_SIX_ARGS 1
#endif
#endif

// --- compile-time feature probes (reported via stderr on success) ----------
#ifdef HAVE_TDM
#ifdef TDM_SIX_ARGS
#warning "CDNA5 probe: tensor_load_to_lds AVAILABLE (six-arg variant)"
#else
#warning "CDNA5 probe: tensor_load_to_lds AVAILABLE (five-arg variant)"
#endif
#else
#warning "CDNA5 probe: tensor_load_to_lds NOT available - async fallback in attention"
#endif
#if !__has_builtin(__builtin_amdgcn_global_load_async_to_lds_b128)
#warning "CDNA5 probe: global_load_async_to_lds_b128 builtin NOT available - inline asm"
#endif
#if !__has_builtin(__builtin_amdgcn_cvt_pk_bf16_f32)
#warning "CDNA5 probe: cvt_pk_bf16_f32 builtin NOT available - manual pack"
#endif

#ifdef HAVE_TDM
__device__ __forceinline__ i32x8 tdm_group1(unsigned tdim0, unsigned tdim1,
                                            unsigned til0, unsigned til1,
                                            unsigned long long stride0) {
    i32x8 g1;
    g1[0] = (int)((1u << 16)        // data_size = 2 bytes
                | (1u << 20)        // pad_enable
                | (4u << 22)        // pad_interval: 32 DWORDs
                | (3u << 25));      // pad_amount: 4 DWORDs
    g1[1] = (int)((tdim0 & 0xffffu) << 16);
    g1[2] = (int)(((tdim0 >> 16) & 0xffffu) | ((tdim1 & 0xffffu) << 16));
    g1[3] = (int)(((tdim1 >> 16) & 0xffffu) | ((til0 & 0xffffu) << 16));
    g1[4] = (int)(til1 & 0xffffu);                     // tile_dim2 = 0
    g1[5] = (int)(unsigned)(stride0 & 0xffffffffull);
    g1[6] = (int)(unsigned)((stride0 >> 32) & 0xffffull);
    g1[7] = 0;
    return g1;
}

__device__ __forceinline__ void tdm_issue(unsigned lds_off, unsigned long long ga,
                                          const i32x8& g1) {
    u32x4 g0;
    g0[0] = 1u;                                       // count=1, user mode
    g0[1] = lds_off;
    g0[2] = (unsigned)(ga & 0xffffffffull);
    g0[3] = (unsigned)((ga >> 32) & 0x1ffffffull) | 0x80000000u;  // type=2
    i32x4 z4 = {0, 0, 0, 0};
#ifdef TDM_SIX_ARGS
    i32x8 z8 = {0, 0, 0, 0, 0, 0, 0, 0};
    __builtin_amdgcn_tensor_load_to_lds(g0, g1, z4, z4, z8, 0);
#else
    __builtin_amdgcn_tensor_load_to_lds(g0, g1, z4, z4, 0);
#endif
}
#endif

// ---------------------------------------------------------------------------
// Problem constants
// ---------------------------------------------------------------------------
#define BATCH   2
#define NTOT    8192
#define DIM     512
#define HEADS   8
#define DHEAD   64
#define NTOK    1024
#define MROWS   (BATCH*NTOT)     // 16384
// d^-0.5 * log2(e): softmax runs in base-2 domain
#define QK_SCALE_LOG2E 0.18033688011112042f

// ---------------------------------------------------------------------------
// fp32 -> bf16 conversion (n multiple of 4)
// ---------------------------------------------------------------------------
__global__ void cvt_f32_bf16(const float* __restrict__ src,
                             unsigned short* __restrict__ dst, int n) {
    int i = (blockIdx.x * blockDim.x + threadIdx.x) * 4;
    if (i >= n) return;
    float4 f = *(const float4*)(src + i);
    uint2 o;
    o.x = pack_bf16(f.x, f.y);
    o.y = pack_bf16(f.z, f.w);
    *(uint2*)(dst + i) = o;
}

// fp32 [K][N] -> bf16 [N][K] (one-time weight transpose)
__global__ void cvt_transpose(const float* __restrict__ src,
                              unsigned short* __restrict__ dst, int K, int N) {
    int n = blockIdx.x * 16 + threadIdx.x;
    int k = blockIdx.y * 16 + threadIdx.y;
    if (n < N && k < K) dst[(size_t)n * K + k] = f2bf(src[(size_t)k * N + n]);
}

// ---------------------------------------------------------------------------
// QKV projection: [16384,512] x Wt[1536,512] -> q/k bf16 [bh][8192][64],
// v bf16 transposed [bh][64][8192]. Block tile 256x128 (8 waves, 64x64 each:
// 16 WMMA per 16 ds_load_b128), double-buffered async LDS.
// grid = (12, 64): bN 0..3 q, 4..7 k, 8..11 v.
// ---------------------------------------------------------------------------
__global__ void __launch_bounds__(256)
qkv_gemm(const unsigned short* __restrict__ xb,    // [16384][512]
         const unsigned short* __restrict__ wbT,   // [1536][512] (N-major)
         unsigned short* __restrict__ qb,
         unsigned short* __restrict__ kbuf,
         unsigned short* __restrict__ vT) {
    __shared__ __attribute__((aligned(16))) unsigned short Alds[2][256 * 40];
    __shared__ __attribute__((aligned(16))) unsigned short Blds[2][128 * 40];

    const int tid   = threadIdx.x;
    const int lane  = tid & 31;
    const int wid   = tid >> 5;
    const int wm    = wid >> 1;       // 0..3 (64-row strip)
    const int wn    = wid & 1;        // 0..1 (64-col strip)
    const int l16   = lane & 15;
    const int lhalf = lane >> 4;
    const int bN    = blockIdx.x;
    const int bM    = blockIdx.y;

    v8f acc[4][4];
    v8f zero = {};
    #pragma unroll
    for (int mi = 0; mi < 4; ++mi)
        #pragma unroll
        for (int ni = 0; ni < 4; ++ni) acc[mi][ni] = zero;

    auto load_tiles = [&](int k0, int buf) {
        #pragma unroll
        for (int t = 0; t < 4; ++t) {            // A: 256x32 = 1024 x 16B
            int idx = tid + t * 256;
            int row = idx >> 2;
            int cu  = idx & 3;
            async_copy16(&Alds[buf][row * 40 + cu * 8],
                         xb + (size_t)(bM * 256 + row) * DIM + k0 + cu * 8);
        }
        #pragma unroll
        for (int t = 0; t < 2; ++t) {            // B: 128x32 = 512 x 16B
            int idx = tid + t * 256;
            int row = idx >> 2;
            int cu  = idx & 3;
            async_copy16(&Blds[buf][row * 40 + cu * 8],
                         wbT + (size_t)(bN * 128 + row) * DIM + k0 + cu * 8);
        }
    };

    load_tiles(0, 0);
    wait_async0();
    __syncthreads();

    #pragma unroll 2
    for (int kt = 0; kt < 16; ++kt) {
        const int buf = kt & 1;
        if (kt + 1 < 16) load_tiles((kt + 1) * 32, buf ^ 1);

        FragAB bfrag[4];
        #pragma unroll
        for (int ni = 0; ni < 4; ++ni) {
            int n  = wn * 64 + ni * 16 + l16;
            int kb = lhalf * 8;
            bfrag[ni].u[0] = *(const uint4*)(&Blds[buf][n * 40 + kb]);
            bfrag[ni].u[1] = *(const uint4*)(&Blds[buf][n * 40 + kb + 16]);
        }
        #pragma unroll
        for (int mi = 0; mi < 4; ++mi) {
            FragAB afrag;
            int row = wm * 64 + mi * 16 + l16;
            int kb  = lhalf * 8;
            afrag.u[0] = *(const uint4*)(&Alds[buf][row * 40 + kb]);
            afrag.u[1] = *(const uint4*)(&Alds[buf][row * 40 + 16 + kb]);
            #pragma unroll
            for (int ni = 0; ni < 4; ++ni)
                acc[mi][ni] = wmma_bf16(afrag, bfrag[ni], acc[mi][ni]);
        }

        wait_async0();
        __syncthreads();
    }

    if (bN < 8) {
        // q (scaled by d^-0.5*log2e) and k: [bh][Ntot][64] scattered b16 stores
        unsigned short* dst = (bN < 4) ? qb : kbuf;
        const float sc = (bN < 4) ? QK_SCALE_LOG2E : 1.0f;
        #pragma unroll
        for (int mi = 0; mi < 4; ++mi)
            #pragma unroll
            for (int ni = 0; ni < 4; ++ni)
                #pragma unroll
                for (int r = 0; r < 8; ++r) {
                    int grow = bM * 256 + wm * 64 + mi * 16 + r + lhalf * 8;
                    int gcol = bN * 128 + wn * 64 + ni * 16 + l16;
                    int b  = grow >> 13;
                    int tt = grow & 8191;
                    int c  = gcol & 511;
                    dst[((size_t)(b * HEADS + (c >> 6)) * NTOT + tt) * DHEAD + (c & 63)] =
                        f2bf(acc[mi][ni][r] * sc);
                }
    } else {
        // v: transposed [bh][d][Ntot]; 8 consecutive tokens per accumulator
        // column -> one b128 store
        #pragma unroll
        for (int mi = 0; mi < 4; ++mi)
            #pragma unroll
            for (int ni = 0; ni < 4; ++ni) {
                const v8f& a = acc[mi][ni];
                uint4 pk;
                pk.x = pack_bf16(a[0], a[1]);
                pk.y = pack_bf16(a[2], a[3]);
                pk.z = pack_bf16(a[4], a[5]);
                pk.w = pack_bf16(a[6], a[7]);
                int grow0 = bM * 256 + wm * 64 + mi * 16 + lhalf * 8;  // mult of 8
                int gcol  = bN * 128 + wn * 64 + ni * 16 + l16;
                int b   = grow0 >> 13;
                int tt0 = grow0 & 8191;
                int c   = gcol & 511;
                *(uint4*)&vT[((size_t)(b * HEADS + (c >> 6)) * DHEAD + (c & 63)) * NTOT + tt0] = pk;
            }
    }
}

// ---------------------------------------------------------------------------
// Flash attention over axial blocks: 128 blocks of [1024, 64].
// grid = (8, 128). 8 waves/WG, wave owns 16 Q rows; KV chunks of 64 are
// double-buffered via Tensor Data Mover (wave 0, TENSORcnt); softmax in
// base-2 domain (Q pre-scaled by d^-0.5*log2e).
// ---------------------------------------------------------------------------
__global__ void __launch_bounds__(256)
axial_attn(const unsigned short* __restrict__ qb,   // [bh][8192][64]
           const unsigned short* __restrict__ kbuf, // [bh][8192][64]
           const unsigned short* __restrict__ vT,   // [bh][64][8192]
           unsigned short* __restrict__ ob) {
    __shared__ __attribute__((aligned(16))) unsigned short Klds[2][64 * 72];  // [kv][d]
    __shared__ __attribute__((aligned(16))) unsigned short Vlds[2][64 * 72];  // [d][kv]
    __shared__ __attribute__((aligned(16))) unsigned short Plds[8 * 16 * 72];

    const int tid   = threadIdx.x;
    const int lane  = tid & 31;
    const int wid   = tid >> 5;
    const int l16   = lane & 15;
    const int lhalf = lane >> 4;
    const int rt    = blockIdx.x;
    const int bb    = blockIdx.y;
    const int bh    = bb >> 3;
    const int ff    = bb & 7;
    const size_t base  = ((size_t)bh * NTOT + (size_t)ff * NTOK) * DHEAD;   // q/k
    const size_t vbase = (size_t)bh * DHEAD * NTOT + (size_t)ff * NTOK;     // vT

    // Q A-fragments in registers for the whole kernel
    FragAB qf[2];
    {
        int row = rt * 128 + wid * 16 + l16;
        int kb  = lhalf * 8;
        #pragma unroll
        for (int ks = 0; ks < 2; ++ks) {
            const unsigned short* p = qb + base + (size_t)row * DHEAD + ks * 32;
            qf[ks].u[0] = *(const uint4*)(p + kb);
            qf[ks].u[1] = *(const uint4*)(p + 16 + kb);
        }
    }

#ifdef HAVE_TDM
    const i32x8 g1k = tdm_group1(64, 1u << 20, 64, 64, 64ull);
    const i32x8 g1v = tdm_group1(64, 1u << 20, 64, 64, (unsigned long long)NTOT);
    const unsigned long long gaK = (unsigned long long)(uintptr_t)(kbuf + base);
    const unsigned long long gaV = (unsigned long long)(uintptr_t)(vT + vbase);
    const unsigned ldsK0 = (unsigned)(uintptr_t)&Klds[0][0];
    const unsigned ldsK1 = (unsigned)(uintptr_t)&Klds[1][0];
    const unsigned ldsV0 = (unsigned)(uintptr_t)&Vlds[0][0];
    const unsigned ldsV1 = (unsigned)(uintptr_t)&Vlds[1][0];
#endif

    auto issue_tiles = [&](int kc, int buf) {
#ifdef HAVE_TDM
        if (wid == 0) {
            tdm_issue(buf ? ldsK1 : ldsK0, gaK + (unsigned long long)kc * (64 * DHEAD * 2), g1k);
            tdm_issue(buf ? ldsV1 : ldsV0, gaV + (unsigned long long)kc * (64 * 2), g1v);
        }
#else
        const unsigned short* kg = kbuf + base + (size_t)kc * 64 * DHEAD;
        const unsigned short* vg = vT + vbase + (size_t)kc * 64;
        #pragma unroll
        for (int t = 0; t < 2; ++t) {
            int idx = tid + t * 256;
            int row = idx >> 3;
            int cu  = idx & 7;
            async_copy16(&Klds[buf][row * 72 + cu * 8], kg + (size_t)row * DHEAD + cu * 8);
            async_copy16(&Vlds[buf][row * 72 + cu * 8], vg + (size_t)row * NTOT + cu * 8);
        }
#endif
    };
    auto wait_tiles = [&]() {
#ifdef HAVE_TDM
        __builtin_amdgcn_s_wait_tensorcnt(0);
#else
        wait_async0();
#endif
    };

    float m_i[8], l_i[8];
    #pragma unroll
    for (int r = 0; r < 8; ++r) { m_i[r] = -3.0e38f; l_i[r] = 0.0f; }
    v8f zero = {};
    v8f accO[4];
    #pragma unroll
    for (int di = 0; di < 4; ++di) accO[di] = zero;

    unsigned short* pw = Plds + wid * 16 * 72;

    issue_tiles(0, 0);
    wait_tiles();
    __syncthreads();

    #pragma unroll 2
    for (int kc = 0; kc < NTOK / 64; ++kc) {
        const int buf = kc & 1;
        if (kc + 1 < NTOK / 64) issue_tiles(kc + 1, buf ^ 1);

        // S = Q * K^T (16 x 64), log2-scaled
        v8f accS[4];
        #pragma unroll
        for (int ni = 0; ni < 4; ++ni) accS[ni] = zero;
        #pragma unroll
        for (int ni = 0; ni < 4; ++ni) {
            int n = ni * 16 + l16;
            #pragma unroll
            for (int ks = 0; ks < 2; ++ks) {
                FragAB bf;
                const unsigned short* p = &Klds[buf][n * 72 + ks * 32 + lhalf * 8];
                bf.u[0] = *(const uint4*)(p);
                bf.u[1] = *(const uint4*)(p + 16);
                accS[ni] = wmma_bf16(qf[ks], bf, accS[ni]);
            }
        }

        // Online softmax in base-2 (rows per-lane as elements r)
        #pragma unroll
        for (int r = 0; r < 8; ++r) {
            float mx = accS[0][r];
            mx = fmaxf(mx, accS[1][r]);
            mx = fmaxf(mx, accS[2][r]);
            mx = fmaxf(mx, accS[3][r]);
            mx = fmaxf(mx, __shfl_xor(mx, 1, 32));
            mx = fmaxf(mx, __shfl_xor(mx, 2, 32));
            mx = fmaxf(mx, __shfl_xor(mx, 4, 32));
            mx = fmaxf(mx, __shfl_xor(mx, 8, 32));
            float mnew = fmaxf(m_i[r], mx);
            float sc   = fast_exp2(m_i[r] - mnew);
            l_i[r] = l_i[r] * sc;
            #pragma unroll
            for (int di = 0; di < 4; ++di) accO[di][r] = accO[di][r] * sc;
            float rs = 0.0f;
            #pragma unroll
            for (int ni = 0; ni < 4; ++ni) {
                float pv = fast_exp2(accS[ni][r] - mnew);
                accS[ni][r] = pv;
                rs += pv;
            }
            rs += __shfl_xor(rs, 1, 32);
            rs += __shfl_xor(rs, 2, 32);
            rs += __shfl_xor(rs, 4, 32);
            rs += __shfl_xor(rs, 8, 32);
            l_i[r] += rs;
            m_i[r] = mnew;
        }

        // P: C-layout -> A-layout via wave-private LDS
        #pragma unroll
        for (int ni = 0; ni < 4; ++ni)
            #pragma unroll
            for (int r = 0; r < 8; ++r)
                pw[(r + lhalf * 8) * 72 + ni * 16 + l16] = f2bf(accS[ni][r]);
        asm volatile("s_wait_dscnt 0x0" ::: "memory");   // DS in-order per wave

        // O += P * V (16 x 64)
        #pragma unroll
        for (int ks = 0; ks < 2; ++ks) {
            FragAB pf;
            const unsigned short* pp = pw + l16 * 72 + ks * 32;
            pf.u[0] = *(const uint4*)(pp + lhalf * 8);
            pf.u[1] = *(const uint4*)(pp + 16 + lhalf * 8);
            #pragma unroll
            for (int di = 0; di < 4; ++di) {
                FragAB vf;
                const unsigned short* vp = &Vlds[buf][(di * 16 + l16) * 72 + ks * 32 + lhalf * 8];
                vf.u[0] = *(const uint4*)(vp);
                vf.u[1] = *(const uint4*)(vp + 16);
                accO[di] = wmma_bf16(pf, vf, accO[di]);
            }
        }

        wait_tiles();
        __syncthreads();
    }

    const int b  = bh >> 3;
    const int hh = bh & 7;
    #pragma unroll
    for (int di = 0; di < 4; ++di)
        #pragma unroll
        for (int r = 0; r < 8; ++r) {
            int j   = rt * 128 + wid * 16 + r + lhalf * 8;
            int t   = ff * NTOK + j;
            int col = di * 16 + l16;
            float ov = accO[di][r] / l_i[r];
            ob[((size_t)b * NTOT + t) * DIM + hh * DHEAD + col] = f2bf(ov);
        }
}

// ---------------------------------------------------------------------------
// Output projection: [16384,512] x WoutT[512,512] + bias -> fp32.
// Block tile 256x128, same async double-buffered structure. grid = (4, 64).
// ---------------------------------------------------------------------------
__global__ void __launch_bounds__(256)
out_gemm(const unsigned short* __restrict__ ab,    // [16384][512]
         const unsigned short* __restrict__ wbT,   // [512][512] (N-major)
         const float* __restrict__ bias,
         float* __restrict__ out) {
    __shared__ __attribute__((aligned(16))) unsigned short Alds[2][256 * 40];
    __shared__ __attribute__((aligned(16))) unsigned short Blds[2][128 * 40];

    const int tid   = threadIdx.x;
    const int lane  = tid & 31;
    const int wid   = tid >> 5;
    const int wm    = wid >> 1;
    const int wn    = wid & 1;
    const int l16   = lane & 15;
    const int lhalf = lane >> 4;
    const int bN    = blockIdx.x;
    const int bM    = blockIdx.y;

    v8f acc[4][4];
    v8f zero = {};
    #pragma unroll
    for (int mi = 0; mi < 4; ++mi)
        #pragma unroll
        for (int ni = 0; ni < 4; ++ni) acc[mi][ni] = zero;

    auto load_tiles = [&](int k0, int buf) {
        #pragma unroll
        for (int t = 0; t < 4; ++t) {
            int idx = tid + t * 256;
            int row = idx >> 2;
            int cu  = idx & 3;
            async_copy16(&Alds[buf][row * 40 + cu * 8],
                         ab + (size_t)(bM * 256 + row) * DIM + k0 + cu * 8);
        }
        #pragma unroll
        for (int t = 0; t < 2; ++t) {
            int idx = tid + t * 256;
            int row = idx >> 2;
            int cu  = idx & 3;
            async_copy16(&Blds[buf][row * 40 + cu * 8],
                         wbT + (size_t)(bN * 128 + row) * DIM + k0 + cu * 8);
        }
    };

    load_tiles(0, 0);
    wait_async0();
    __syncthreads();

    #pragma unroll 2
    for (int kt = 0; kt < 16; ++kt) {
        const int buf = kt & 1;
        if (kt + 1 < 16) load_tiles((kt + 1) * 32, buf ^ 1);

        FragAB bfrag[4];
        #pragma unroll
        for (int ni = 0; ni < 4; ++ni) {
            int n  = wn * 64 + ni * 16 + l16;
            int kb = lhalf * 8;
            bfrag[ni].u[0] = *(const uint4*)(&Blds[buf][n * 40 + kb]);
            bfrag[ni].u[1] = *(const uint4*)(&Blds[buf][n * 40 + kb + 16]);
        }
        #pragma unroll
        for (int mi = 0; mi < 4; ++mi) {
            FragAB afrag;
            int row = wm * 64 + mi * 16 + l16;
            int kb  = lhalf * 8;
            afrag.u[0] = *(const uint4*)(&Alds[buf][row * 40 + kb]);
            afrag.u[1] = *(const uint4*)(&Alds[buf][row * 40 + 16 + kb]);
            #pragma unroll
            for (int ni = 0; ni < 4; ++ni)
                acc[mi][ni] = wmma_bf16(afrag, bfrag[ni], acc[mi][ni]);
        }

        wait_async0();
        __syncthreads();
    }

    #pragma unroll
    for (int mi = 0; mi < 4; ++mi)
        #pragma unroll
        for (int ni = 0; ni < 4; ++ni)
            #pragma unroll
            for (int r = 0; r < 8; ++r) {
                int grow = bM * 256 + wm * 64 + mi * 16 + r + lhalf * 8;
                int gcol = bN * 128 + wn * 64 + ni * 16 + l16;
                out[(size_t)grow * DIM + gcol] = acc[mi][ni][r] + bias[gcol];
            }
}

// ---------------------------------------------------------------------------
// Host entry
// ---------------------------------------------------------------------------
extern "C" void kernel_launch(void* const* d_in, const int* in_sizes, int n_in,
                              void* d_out, int out_size, void* d_ws, size_t ws_size,
                              hipStream_t stream) {
    const float* x     = (const float*)d_in[0];   // [2,8192,512]
    const float* W_qkv = (const float*)d_in[1];   // [512,1536]
    const float* W_out = (const float*)d_in[2];   // [512,512]
    const float* b_out = (const float*)d_in[3];   // [512]
    (void)in_sizes; (void)n_in; (void)ws_size;

    char* ws = (char*)d_ws;
    size_t off = 0;
    auto alloc = [&](size_t bytes) -> char* {
        char* p = ws + off;
        off += (bytes + 255) & ~(size_t)255;
        return p;
    };
    unsigned short* xb     = (unsigned short*)alloc((size_t)MROWS * DIM * 2);
    unsigned short* wqkvT  = (unsigned short*)alloc((size_t)1536 * DIM * 2);
    unsigned short* woutT  = (unsigned short*)alloc((size_t)DIM * DIM * 2);
    unsigned short* qb     = (unsigned short*)alloc((size_t)16 * NTOT * DHEAD * 2);
    unsigned short* kbuf   = (unsigned short*)alloc((size_t)16 * NTOT * DHEAD * 2);
    unsigned short* vT     = (unsigned short*)alloc((size_t)16 * NTOT * DHEAD * 2);
    unsigned short* obuf   = (unsigned short*)alloc((size_t)MROWS * DIM * 2);

    cvt_f32_bf16<<<(MROWS * DIM) / 1024, 256, 0, stream>>>(x, xb, MROWS * DIM);
    cvt_transpose<<<dim3(1536 / 16, DIM / 16), dim3(16, 16), 0, stream>>>(W_qkv, wqkvT, DIM, 1536);
    cvt_transpose<<<dim3(DIM / 16, DIM / 16), dim3(16, 16), 0, stream>>>(W_out, woutT, DIM, DIM);

    qkv_gemm<<<dim3(12, 64), 256, 0, stream>>>(xb, wqkvT, qb, kbuf, vT);
    axial_attn<<<dim3(8, 128), 256, 0, stream>>>(qb, kbuf, vT, obuf);
    out_gemm<<<dim3(4, 64), 256, 0, stream>>>(obuf, woutT, b_out, (float*)d_out);
}
